// SparseGraphNetwork_54365696033258
// MI455X (gfx1250) — compile-verified, matrix operand order checked by reading, not run on hardware
//
#include <hip/hip_runtime.h>

// ---------------------------------------------------------------------------
// GraphSAGE (3x SAGEConv + skip, + linear head) for MI455X / gfx1250.
//  - Edge aggregation: f32 global atomics (agg = 25.6MB, L2-resident).
//  - GEMMs: bf16 V_WMMA_F32_16X16X32_BF16, f32 accumulate.
//    Weights pre-packed on device into per-lane WMMA B-fragment order,
//    staged per-block into LDS (64KB of the WGP's 320KB), fragments fed by
//    ds_load_b128 pairs; A fragments held in VGPRs across the tile sweep.
//  - Stores: wave-uniform fast path (unguarded) for fully in-range strips.
// ---------------------------------------------------------------------------

typedef __attribute__((ext_vector_type(16))) __bf16 v16bf;
typedef __attribute__((ext_vector_type(8)))  float  v8f;
typedef unsigned short u16;
typedef unsigned int   u32;

#define NNODES 50000
#define NEDGES 800000
#define KDIM   128

__device__ __forceinline__ u16 f2bf(float f) {
  u32 u = __float_as_uint(f);
  u32 r = (u + 0x7FFFu + ((u >> 16) & 1u)) >> 16;   // round-to-nearest-even
  return (u16)r;
}

__global__ void zero_f32(float* __restrict__ p, int n) {
  int t = blockIdx.x * blockDim.x + threadIdx.x;
  if (t < n) p[t] = 0.f;
}

// Pack W (row-major [K=128][ncols], f32, optionally W1+W2) into the per-lane
// bf16 WMMA B-fragment order: pk[((ct*4+kki)*32+lane)*16 + s], where
// n = ct*16 + (lane&15), hs = lane>>4, K = kki*32 + hs*8 + (s<8 ? s : s+8).
__global__ void pack_weight(const float* __restrict__ W1, const float* __restrict__ W2,
                            u16* __restrict__ pk, int ncols) {
  int tid = blockIdx.x * blockDim.x + threadIdx.x;
  int total = (ncols >> 4) * 4 * 32 * 16;
  if (tid >= total) return;
  int s    = tid & 15;
  int lane = (tid >> 4) & 31;
  int kki  = (tid >> 9) & 3;
  int ct   = tid >> 11;
  int hs = lane >> 4;
  int n  = ct * 16 + (lane & 15);
  int K  = kki * 32 + hs * 8 + (s < 8 ? s : s + 8);
  float v = W1[K * ncols + n];
  if (W2) v += W2[K * ncols + n];
  pk[tid] = f2bf(v);
}

__global__ void add_bias(const float* __restrict__ a, const float* __restrict__ b,
                         float* __restrict__ o, int n) {
  int t = blockIdx.x * blockDim.x + threadIdx.x;
  if (t < n) o[t] = a[t] + b[t];
}

// One thread per (edge, 4-channel chunk): float4 gather + 4 f32 atomic adds.
__global__ void scatter_edges(const int* __restrict__ src, const int* __restrict__ dst,
                              const float* __restrict__ x, float* __restrict__ agg,
                              float* __restrict__ deg, int ne) {
  int tid = blockIdx.x * blockDim.x + threadIdx.x;
  int e  = tid >> 5;
  int c4 = tid & 31;
  if (e >= ne) return;
  int s = src[e], d = dst[e];
  const float4 v = *reinterpret_cast<const float4*>(x + (size_t)s * KDIM + c4 * 4);
  float* ap = agg + (size_t)d * KDIM + c4 * 4;
  atomicAdd(ap + 0, v.x);
  atomicAdd(ap + 1, v.y);
  atomicAdd(ap + 2, v.z);
  atomicAdd(ap + 3, v.w);
  if (c4 == 0) atomicAdd(deg + d, 1.0f);
}

// mean = agg / max(deg,1) -> bf16 ; also convert current features to bf16.
__global__ void mean_cvt(const float* __restrict__ agg, const float* __restrict__ deg,
                         const float* __restrict__ xcur,
                         u16* __restrict__ meanb, u16* __restrict__ xb, int n) {
  int t = blockIdx.x * blockDim.x + threadIdx.x;
  if (t >= n) return;
  int node = t >> 7;                      // /128
  float dg = deg[node];
  dg = dg > 1.f ? dg : 1.f;
  meanb[t] = f2bf(agg[t] / dg);
  xb[t]    = f2bf(xcur[t]);
}

__global__ void cvt_bf16(const float* __restrict__ x, u16* __restrict__ o, int n) {
  int t = blockIdx.x * blockDim.x + threadIdx.x;
  if (t < n) o[t] = f2bf(x[t]);
}

// A fragment (16x32 bf16, ISA layout): lane m = lane&15 owns row, hs = lane>>4.
// elems 0..7  -> K = kk + hs*8 + 0..7     (one aligned uint4)
// elems 8..15 -> K = kk + 16 + hs*8 + 0..7 (one aligned uint4)
__device__ __forceinline__ v16bf load_a_frag(const u16* __restrict__ rowp, int kk, int hs) {
  union { uint4 u[2]; v16bf v; } t;
  t.u[0] = *reinterpret_cast<const uint4*>(rowp + kk + hs * 8);
  t.u[1] = *reinterpret_cast<const uint4*>(rowp + kk + 16 + hs * 8);
  return t.v;
}

// B fragment from LDS-staged packed weights: 32 contiguous bytes per lane.
__device__ __forceinline__ v16bf load_b_frag_lds(const u16* sp, int ct, int kki, int lane) {
  union { uint4 u[2]; v16bf v; } t;
  const uint4* p = reinterpret_cast<const uint4*>(sp + ((((ct * 4 + kki) * 32) + lane) << 4));
  t.u[0] = p[0];
  t.u[1] = p[1];
  return t.v;
}

// out[nrows x ncols] = A1 @ B1 (+ A2 @ B2) + bias, optional ReLU. K = 128.
// 256 threads = 8 waves; each wave owns a 16-row strip, sweeps ncols/16 tiles.
__global__ __launch_bounds__(256) void gemm_wmma(
    const u16* __restrict__ A1, const u16* __restrict__ B1,
    const u16* __restrict__ A2, const u16* __restrict__ B2,
    const float* __restrict__ bias, float* __restrict__ out,
    int nrows, int ncols, int relu) {
  // --- cooperative stage of packed weights into LDS (all threads, pre-exit) ---
  __shared__ __align__(16) u16 smem[2 * KDIM * KDIM];   // 64KB max
  const int nElem  = ncols * KDIM;                      // packed elements per matrix
  const int total4 = nElem >> 3;                        // uint4 chunks
  {
    const uint4* g1 = reinterpret_cast<const uint4*>(B1);
    uint4* s4 = reinterpret_cast<uint4*>(smem);
    for (int i = threadIdx.x; i < total4; i += 256) s4[i] = g1[i];
    if (A2) {
      const uint4* g2 = reinterpret_cast<const uint4*>(B2);
      uint4* s42 = reinterpret_cast<uint4*>(smem + nElem);
      for (int i = threadIdx.x; i < total4; i += 256) s42[i] = g2[i];
    }
  }
  __syncthreads();   // every wave reaches this before any exits (no barrier hang)

  const int lane = threadIdx.x & 31;
  const int wave = threadIdx.x >> 5;
  const int rowBase = blockIdx.x * 128 + wave * 16;
  if (rowBase >= nrows) return;
  const bool fullStrip = (rowBase + 16) <= nrows;       // wave-uniform fast path
  const int hs = lane >> 4;
  int arow = rowBase + (lane & 15);
  if (arow >= nrows) arow = nrows - 1;                  // clamp loads; stores guarded

  v16bf a1[4], a2[4];
  const u16* r1 = A1 + (size_t)arow * KDIM;
#pragma unroll
  for (int k = 0; k < 4; ++k) a1[k] = load_a_frag(r1, k * 32, hs);
  if (A2) {
    const u16* r2 = A2 + (size_t)arow * KDIM;
#pragma unroll
    for (int k = 0; k < 4; ++k) a2[k] = load_a_frag(r2, k * 32, hs);
  }

  const u16* sB1 = smem;
  const u16* sB2 = smem + nElem;
  const int ntiles = ncols >> 4;
  for (int ct = 0; ct < ntiles; ++ct) {
    v8f acc = {0.f, 0.f, 0.f, 0.f, 0.f, 0.f, 0.f, 0.f};
    v16bf b[4];
#pragma unroll
    for (int k = 0; k < 4; ++k) b[k] = load_b_frag_lds(sB1, ct, k, lane);
#pragma unroll
    for (int k = 0; k < 4; ++k)
      acc = __builtin_amdgcn_wmma_f32_16x16x32_bf16(false, a1[k], false, b[k],
                                                    (short)0, acc, false, false);
    if (A2) {
#pragma unroll
      for (int k = 0; k < 4; ++k) b[k] = load_b_frag_lds(sB2, ct, k, lane);
#pragma unroll
      for (int k = 0; k < 4; ++k)
        acc = __builtin_amdgcn_wmma_f32_16x16x32_bf16(false, a2[k], false, b[k],
                                                      (short)0, acc, false, false);
    }
    const int n  = ct * 16 + (lane & 15);
    const float bv = bias ? bias[n] : 0.f;
    const int m0 = (lane < 16) ? 0 : 8;                 // C/D ISA layout
    float* op = out + (size_t)(rowBase + m0) * ncols + n;
    if (fullStrip) {                                    // common case: no guards
      if (relu) {
#pragma unroll
        for (int j = 0; j < 8; ++j) {
          float v = acc[j] + bv;
          op[(size_t)j * ncols] = v > 0.f ? v : 0.f;
        }
      } else {
#pragma unroll
        for (int j = 0; j < 8; ++j)
          op[(size_t)j * ncols] = acc[j] + bv;
      }
    } else {                                            // last partial block only
#pragma unroll
      for (int j = 0; j < 8; ++j) {
        int row = rowBase + m0 + j;
        if (row < nrows) {
          float v = acc[j] + bv;
          if (relu) v = v > 0.f ? v : 0.f;
          out[(size_t)row * ncols + n] = v;
        }
      }
    }
  }
}

extern "C" void kernel_launch(void* const* d_in, const int* in_sizes, int n_in,
                              void* d_out, int out_size, void* d_ws, size_t ws_size,
                              hipStream_t stream) {
  (void)in_sizes; (void)n_in; (void)out_size; (void)ws_size;
  const float* x   = (const float*)d_in[0];
  const int*   ei  = (const int*)d_in[1];        // [2, E]; per-harness integer rule
  const int*   srcv = ei;
  const int*   dstv = ei + NEDGES;
  const float* Wl[3] = {(const float*)d_in[2], (const float*)d_in[7],  (const float*)d_in[12]};
  const float* bl[3] = {(const float*)d_in[3], (const float*)d_in[8],  (const float*)d_in[13]};
  const float* Wr[3] = {(const float*)d_in[4], (const float*)d_in[9],  (const float*)d_in[14]};
  const float* Ws[3] = {(const float*)d_in[5], (const float*)d_in[10], (const float*)d_in[15]};
  const float* bs[3] = {(const float*)d_in[6], (const float*)d_in[11], (const float*)d_in[16]};
  const float* Wh = (const float*)d_in[17];
  const float* bh = (const float*)d_in[18];

  char* ws = (char*)d_ws;
  size_t off = 0;
  auto alloc = [&](size_t bytes) -> char* {
    char* p = ws + off;
    off = (off + bytes + 255) & ~(size_t)255;
    return p;
  };

  const size_t NF = (size_t)NNODES * KDIM;
  float* agg   = (float*)alloc(NF * 4);
  float* deg   = (float*)alloc((size_t)NNODES * 4);
  float* y0    = (float*)alloc(NF * 4);
  float* y1    = (float*)alloc(NF * 4);
  u16*   meanb = (u16*)  alloc(NF * 2);
  u16*   xb    = (u16*)  alloc(NF * 2);
  u16*   pWl[3]; u16* pWc[3]; float* bc[3];
  for (int i = 0; i < 3; ++i) {
    pWl[i] = (u16*)alloc((size_t)KDIM * KDIM * 2);
    pWc[i] = (u16*)alloc((size_t)KDIM * KDIM * 2);
    bc[i]  = (float*)alloc(KDIM * 4);
  }
  u16* pWh = (u16*)alloc((size_t)KDIM * 64 * 2);

  // ---- weight prep (tiny) ----
  for (int i = 0; i < 3; ++i) {
    pack_weight<<<(KDIM * KDIM + 255) / 256, 256, 0, stream>>>(Wl[i], nullptr, pWl[i], KDIM);
    pack_weight<<<(KDIM * KDIM + 255) / 256, 256, 0, stream>>>(Wr[i], Ws[i],   pWc[i], KDIM);
    add_bias<<<1, KDIM, 0, stream>>>(bl[i], bs[i], bc[i], KDIM);
  }
  pack_weight<<<(KDIM * 64 + 255) / 256, 256, 0, stream>>>(Wh, nullptr, pWh, 64);

  // ---- layers ----
  const float* xcur = x;
  float* ping[2] = {y0, y1};
  const int gemmBlocks = (NNODES + 127) / 128;
  for (int i = 0; i < 3; ++i) {
    zero_f32<<<((int)NF + 255) / 256, 256, 0, stream>>>(agg, (int)NF);
    zero_f32<<<(NNODES + 255) / 256, 256, 0, stream>>>(deg, NNODES);
    long long sthreads = (long long)NEDGES * 32;
    scatter_edges<<<(int)((sthreads + 255) / 256), 256, 0, stream>>>(srcv, dstv, xcur, agg, deg, NEDGES);
    mean_cvt<<<((int)NF + 255) / 256, 256, 0, stream>>>(agg, deg, xcur, meanb, xb, (int)NF);
    float* yout = ping[i & 1];
    gemm_wmma<<<gemmBlocks, 256, 0, stream>>>(meanb, pWl[i], xb, pWc[i], bc[i], yout,
                                              NNODES, KDIM, (i != 2) ? 1 : 0);
    xcur = yout;
  }

  // ---- head: [N,128] @ [128,64] + bh ----
  cvt_bf16<<<((int)NF + 255) / 256, 256, 0, stream>>>(xcur, xb, (int)NF);
  gemm_wmma<<<gemmBlocks, 256, 0, stream>>>(xb, pWh, nullptr, nullptr, bh,
                                            (float*)d_out, NNODES, 64, 0);
}
